// GCNModelVAE_6743098655051
// MI455X (gfx1250) — matrix operand order repaired and checked by reading, not compiled, not run on hardware
//
#include <hip/hip_runtime.h>
#include <hip/hip_bf16.h>
#include <math.h>

// ---------------------------------------------------------------------------
// GCN-VAE pipeline for MI455X (gfx1250, wave32, WMMA)
//   1) Xb = bf16(X) padded K 3000->3008 ; W0T = bf16(W0^T) padded
//   2) XW0 = Xb @ W0T^T              (WMMA bf16 -> f32)
//   3) agg1 = segment_sum(ew * XW0[col], row)   (f32 atomics)
//   4) h1b  = bf16(relu(agg1))
//   5) HW1 = h1b @ W1T^T ; HW2 = h1b @ W2T^T    (WMMA)
//   6) aggB/aggU = segment_sum(...)             (f32 atomics)
//   7) latent: sigmoid, Kumaraswamy sample -> z(bf16), kl_d, omega
//   8) recon = sigmoid(z @ z^T)                 (WMMA, sigmoid fused epilogue)
// ---------------------------------------------------------------------------

typedef __bf16 bf16_t;
typedef __attribute__((ext_vector_type(16))) __bf16 v16bf;
typedef __attribute__((ext_vector_type(8)))  __bf16 v8bf;
typedef __attribute__((ext_vector_type(8)))  float  v8f;

#define N_NODES 8192
#define F_IN    3000
#define K1P     3008   // F padded to multiple of 32
#define H1_DIM  512
#define H2_DIM  128
#define E_EDGES 262144
#define EPSV    1e-7f
#define CAPV    0.999999f
#define EULER_C 0.5772f

__device__ __forceinline__ float sigmoidf_(float x) {
    return 1.0f / (1.0f + __expf(-x));
}

__device__ __forceinline__ float digammaf_(float x) {
    // recurrence to x>=6, then asymptotic series
    float r = 0.0f;
    while (x < 6.0f) { r -= 1.0f / x; x += 1.0f; }
    float inv = 1.0f / x, inv2 = inv * inv;
    return r + __logf(x) - 0.5f * inv
           - inv2 * (0.0833333333f - inv2 * (0.0083333333f - inv2 * 0.0039682540f));
}

// ---------------------------------------------------------------------------
// Generic WMMA GEMM: C[M,N] = A[M,K] * BT[N,K]^T   (bf16 in, f32 out)
// 256 threads = 8 waves; block tile 128x128; wave tile 64(M) x 32(N)
// K must be a multiple of 32; M,N multiples of 128 (grid covers exactly).
// ---------------------------------------------------------------------------
__global__ __launch_bounds__(256)
void wmma_gemm_bf16(const bf16_t* __restrict__ A, const bf16_t* __restrict__ BT,
                    float* __restrict__ C, int M, int N, int K, int apply_sigmoid)
{
    const int lane  = threadIdx.x & 31;
    const int wave  = threadIdx.x >> 5;
    const int waveM = wave & 1;            // 2 wave rows
    const int waveN = wave >> 1;           // 4 wave cols
    const int tileM = blockIdx.y * 128 + waveM * 64;
    const int tileN = blockIdx.x * 128 + waveN * 32;
    const int lhalf = lane >> 4;           // 0 | 1
    const int l16   = lane & 15;
    const int kOffA = lhalf * 8;           // A frag: lanes 0-15 K{0..7,16..23}
    const int kOffB = lhalf * 8;           // B frag: lanes 0-15 K{0..7,16..23}

    v8f acc[4][2] = {};

    for (int k0 = 0; k0 < K; k0 += 32) {
        v16bf a[4], b[2];
        #pragma unroll
        for (int mi = 0; mi < 4; ++mi) {
            const bf16_t* p = A + (size_t)(tileM + mi * 16 + l16) * K + k0 + kOffA;
            v8bf lo = *(const v8bf*)p;
            v8bf hi = *(const v8bf*)(p + 16);
            #pragma unroll
            for (int e = 0; e < 8; ++e) { a[mi][e] = lo[e]; a[mi][e + 8] = hi[e]; }
        }
        #pragma unroll
        for (int ni = 0; ni < 2; ++ni) {
            const bf16_t* p = BT + (size_t)(tileN + ni * 16 + l16) * K + k0 + kOffB;
            v8bf lo = *(const v8bf*)p;
            v8bf hi = *(const v8bf*)(p + 16);
            #pragma unroll
            for (int e = 0; e < 8; ++e) { b[ni][e] = lo[e]; b[ni][e + 8] = hi[e]; }
        }
        #pragma unroll
        for (int mi = 0; mi < 4; ++mi)
            #pragma unroll
            for (int ni = 0; ni < 2; ++ni)
                acc[mi][ni] = __builtin_amdgcn_wmma_f32_16x16x32_bf16(
                    false, a[mi], false, b[ni], (short)0, acc[mi][ni], false, false);
    }

    // epilogue: C/D layout -> VGPR r holds row r (lanes 0-15) / r+8 (lanes 16-31)
    #pragma unroll
    for (int mi = 0; mi < 4; ++mi) {
        #pragma unroll
        for (int ni = 0; ni < 2; ++ni) {
            #pragma unroll
            for (int r = 0; r < 8; ++r) {
                int row = tileM + mi * 16 + r + lhalf * 8;
                int col = tileN + ni * 16 + l16;
                float v = acc[mi][ni][r];
                if (apply_sigmoid) v = sigmoidf_(v);
                C[(size_t)row * N + col] = v;
            }
        }
    }
}

// ---------------------------------------------------------------------------
// Elementwise prep kernels
// ---------------------------------------------------------------------------
__global__ void cast_pad_kernel(const float* __restrict__ src, bf16_t* __restrict__ dst,
                                int cols, int colsP, long long total)
{
    long long i = (long long)blockIdx.x * blockDim.x + threadIdx.x;
    if (i >= total) return;
    int rr = (int)(i / colsP);
    int c  = (int)(i % colsP);
    dst[i] = (c < cols) ? (bf16_t)src[(size_t)rr * cols + c] : (bf16_t)0.0f;
}

__global__ void transpose_cast_pad_kernel(const float* __restrict__ W, bf16_t* __restrict__ WT,
                                          int K, int Nc, int Kp, long long total)
{
    // W: [K, Nc] row-major -> WT: [Nc, Kp]  (pad K..Kp with zero)
    long long i = (long long)blockIdx.x * blockDim.x + threadIdx.x;
    if (i >= total) return;
    int n = (int)(i / Kp);
    int k = (int)(i % Kp);
    WT[i] = (k < K) ? (bf16_t)W[(size_t)k * Nc + n] : (bf16_t)0.0f;
}

__global__ void relu_cast_kernel(const float* __restrict__ src, bf16_t* __restrict__ dst, int total)
{
    int i = blockIdx.x * blockDim.x + threadIdx.x;
    if (i < total) dst[i] = (bf16_t)fmaxf(src[i], 0.0f);
}

// ---------------------------------------------------------------------------
// Weighted edge aggregation: out[row[e], :] += ew[e] * h[col[e], :]
// ---------------------------------------------------------------------------
__global__ void edge_agg_kernel(const float* __restrict__ h, const float* __restrict__ ew,
                                const int* __restrict__ rowi, const int* __restrict__ coli,
                                float* __restrict__ out, int H, long long total)
{
    long long idx = (long long)blockIdx.x * blockDim.x + threadIdx.x;
    if (idx >= total) return;
    int e = (int)(idx / H);
    int j = (int)(idx % H);
    float v = ew[e] * h[(size_t)coli[e] * H + j];
    atomicAdd(&out[(size_t)rowi[e] * H + j], v);
}

// ---------------------------------------------------------------------------
// Latent: sigmoid + subjective-logic params + Kumaraswamy sample + KL + omega
// ---------------------------------------------------------------------------
__global__ void latent_kernel(const float* __restrict__ aggB, const float* __restrict__ aggU,
                              const float* __restrict__ uni, const float* __restrict__ a0p,
                              const float* __restrict__ b0p, bf16_t* __restrict__ zb,
                              float* __restrict__ kl_out, float* __restrict__ omega_out,
                              int total)
{
    int i = blockIdx.x * blockDim.x + threadIdx.x;
    if (i >= total) return;

    float belief    = sigmoidf_(aggB[i]);
    float uncertain = sigmoidf_(aggU[i]);

    float b = fminf(belief + EPSV, CAPV);
    float u = fminf(uncertain + EPSV, CAPV);
    float r = 2.0f * b / u + EPSV;
    float d = fminf(fmaxf(1.0f - b - u, 0.0f) + EPSV, CAPV);
    float s = 2.0f * d / u + EPSV;
    float alpha = r + 1.0f + EPSV;   // + Wc*a, Wc=2, a=0.5
    float beta  = s + 1.0f + EPSV;   // + Wc*(1-a)

    // Kumaraswamy inverse-CDF sample -> standard normal quantile
    float un  = 0.001f + 0.998f * uni[i];
    float znb = fminf(1.0f - __powf(un, 1.0f / beta) + EPSV, CAPV);
    float zn  = fminf(__powf(znb, 1.0f / alpha) + EPSV, CAPV);
    float z   = 1.41421356237f * erfinvf(2.0f * zn - 1.0f);
    zb[i] = (bf16_t)z;

    float a0 = a0p[0], b0 = b0p[0];
    float lbeta0 = lgammaf(a0) + lgammaf(b0) - lgammaf(a0 + b0);
    float inv_beta = 1.0f / beta;
    float kl1 = (1.0f - a0 / alpha) * (-EULER_C - digammaf_(beta) - inv_beta)
                + __logf(alpha * beta) + lbeta0 - 1.0f + inv_beta;

    float lgb = lgammaf(beta);
    float ab  = alpha * beta;
    float kl2 = 0.0f;
    #pragma unroll
    for (int m = 1; m <= 10; ++m) {
        float mf = (float)m;
        float t  = lgammaf(mf / alpha) + lgb - lgammaf(mf / alpha + beta);
        kl2 += __expf(t) / (mf + ab);
    }

    kl_out[i]    = kl1 + beta * (b0 - 1.0f) * kl2;
    omega_out[i] = alpha / (alpha + beta);
}

// ---------------------------------------------------------------------------
// Host-side orchestration
// ---------------------------------------------------------------------------
extern "C" void kernel_launch(void* const* d_in, const int* in_sizes, int n_in,
                              void* d_out, int out_size, void* d_ws, size_t ws_size,
                              hipStream_t stream)
{
    (void)in_sizes; (void)n_in; (void)out_size; (void)ws_size;

    const float* features = (const float*)d_in[0];
    const int*   erow     = (const int*)d_in[1];
    const int*   ecol     = (const int*)d_in[2];
    const float* ew       = (const float*)d_in[3];
    const float* uni      = (const float*)d_in[4];
    const float* W0       = (const float*)d_in[5];
    const float* W1       = (const float*)d_in[6];
    const float* W2       = (const float*)d_in[7];
    const float* a0       = (const float*)d_in[8];
    const float* b0       = (const float*)d_in[9];

    char*  ws  = (char*)d_ws;
    size_t off = 0;
    auto carve = [&](size_t bytes) -> void* {
        void* p = ws + off;
        off += (bytes + 255) & ~(size_t)255;
        return p;
    };

    bf16_t* Xb   = (bf16_t*)carve((size_t)N_NODES * K1P * sizeof(bf16_t));
    bf16_t* W0T  = (bf16_t*)carve((size_t)H1_DIM  * K1P * sizeof(bf16_t));
    float*  XW0  = (float*) carve((size_t)N_NODES * H1_DIM * sizeof(float));
    float*  agg1 = (float*) carve((size_t)N_NODES * H1_DIM * sizeof(float));
    bf16_t* h1b  = (bf16_t*)carve((size_t)N_NODES * H1_DIM * sizeof(bf16_t));
    bf16_t* W1T  = (bf16_t*)carve((size_t)H2_DIM  * H1_DIM * sizeof(bf16_t));
    bf16_t* W2T  = (bf16_t*)carve((size_t)H2_DIM  * H1_DIM * sizeof(bf16_t));
    float*  HW1  = (float*) carve((size_t)N_NODES * H2_DIM * sizeof(float));
    float*  HW2  = (float*) carve((size_t)N_NODES * H2_DIM * sizeof(float));
    float*  aggB = (float*) carve((size_t)N_NODES * H2_DIM * sizeof(float));
    float*  aggU = (float*) carve((size_t)N_NODES * H2_DIM * sizeof(float));
    bf16_t* zb   = (bf16_t*)carve((size_t)N_NODES * H2_DIM * sizeof(bf16_t));

    float* recon     = (float*)d_out;
    float* kl_out    = recon  + (size_t)N_NODES * N_NODES;
    float* omega_out = kl_out + (size_t)N_NODES * H2_DIM;

    const int T = 256;
    auto blocks = [&](long long tot) -> unsigned int {
        return (unsigned int)((tot + T - 1) / T);
    };

    // 1) cast features -> bf16, pad K to 3008
    {
        long long tot = (long long)N_NODES * K1P;
        cast_pad_kernel<<<blocks(tot), T, 0, stream>>>(features, Xb, F_IN, K1P, tot);
    }
    // 2) W0^T -> bf16 [H1, K1P]
    {
        long long tot = (long long)H1_DIM * K1P;
        transpose_cast_pad_kernel<<<blocks(tot), T, 0, stream>>>(W0, W0T, F_IN, H1_DIM, K1P, tot);
    }
    // 3) XW0 = Xb @ W0 (WMMA)
    wmma_gemm_bf16<<<dim3(H1_DIM / 128, N_NODES / 128), 256, 0, stream>>>(
        Xb, W0T, XW0, N_NODES, H1_DIM, K1P, 0);

    // 4-5) layer-1 edge aggregation
    hipMemsetAsync(agg1, 0, (size_t)N_NODES * H1_DIM * sizeof(float), stream);
    {
        long long tot = (long long)E_EDGES * H1_DIM;
        edge_agg_kernel<<<blocks(tot), T, 0, stream>>>(XW0, ew, erow, ecol, agg1, H1_DIM, tot);
    }
    // 6) h1 = bf16(relu(agg1))
    relu_cast_kernel<<<blocks((long long)N_NODES * H1_DIM), T, 0, stream>>>(
        agg1, h1b, N_NODES * H1_DIM);

    // 7-8) W1^T, W2^T -> bf16 [H2, H1]
    {
        long long tot = (long long)H2_DIM * H1_DIM;
        transpose_cast_pad_kernel<<<blocks(tot), T, 0, stream>>>(W1, W1T, H1_DIM, H2_DIM, H1_DIM, tot);
        transpose_cast_pad_kernel<<<blocks(tot), T, 0, stream>>>(W2, W2T, H1_DIM, H2_DIM, H1_DIM, tot);
    }
    // 9-10) HW1, HW2 (WMMA)
    wmma_gemm_bf16<<<dim3(H2_DIM / 128, N_NODES / 128), 256, 0, stream>>>(
        h1b, W1T, HW1, N_NODES, H2_DIM, H1_DIM, 0);
    wmma_gemm_bf16<<<dim3(H2_DIM / 128, N_NODES / 128), 256, 0, stream>>>(
        h1b, W2T, HW2, N_NODES, H2_DIM, H1_DIM, 0);

    // 11-12) layer-2 edge aggregation
    hipMemsetAsync(aggB, 0, (size_t)N_NODES * H2_DIM * sizeof(float), stream);
    hipMemsetAsync(aggU, 0, (size_t)N_NODES * H2_DIM * sizeof(float), stream);
    {
        long long tot = (long long)E_EDGES * H2_DIM;
        edge_agg_kernel<<<blocks(tot), T, 0, stream>>>(HW1, ew, erow, ecol, aggB, H2_DIM, tot);
        edge_agg_kernel<<<blocks(tot), T, 0, stream>>>(HW2, ew, erow, ecol, aggU, H2_DIM, tot);
    }

    // 13) latent: z (bf16), kl_d, omega
    latent_kernel<<<blocks((long long)N_NODES * H2_DIM), T, 0, stream>>>(
        aggB, aggU, uni, a0, b0, zb, kl_out, omega_out, N_NODES * H2_DIM);

    // 14) recon = sigmoid(z @ z^T)   (BT == z since B = z^T)
    wmma_gemm_bf16<<<dim3(N_NODES / 128, N_NODES / 128), 256, 0, stream>>>(
        zb, zb, recon, N_NODES, N_NODES, H2_DIM, 1);
}
